// BidirectionalLinearAttention_74096775790883
// MI455X (gfx1250) — matrix operand order repaired and checked by reading, not compiled, not run on hardware
//
#include <hip/hip_runtime.h>
#include <hip/hip_bf16.h>

// ---------------------------------------------------------------------------
// Bidirectional linear attention, MI455X (gfx1250), bf16 WMMA pipeline.
//   B=4, N=8192, D=1024, H=16, d=64.
//   1) convert x, Wq, Wk, Wv, Wo  fp32 -> bf16
//   2) Q  = phi(x Wq^T)  bf16 [32768,1024];  KT/VT = transposed per batch
//   3) kvT = v^T k (fp32 atomics, K split 16 ways); ksum = sum_n k
//   4) kvp[b,h] = bf16 [80][64]: rows 0..63 kvT, row 64 ksum, 65..79 zero
//   5) y = (q @ kvp) with fused divide by (col 64 + eps), bf16
//   6) out = y Wo^T + bo, fp32
// ---------------------------------------------------------------------------

typedef __attribute__((ext_vector_type(16))) __bf16 v16bf;
typedef __attribute__((ext_vector_type(8)))  __bf16 v8bf;
typedef __attribute__((ext_vector_type(8)))  float  v8f;

#define MTOT 32768
#define DDIM 1024
#define NTOK 8192
#define NH   16
#define HD   64
#define KVSPLIT 16

__device__ __forceinline__ unsigned short f2bf(float f) {
  unsigned int u = __float_as_uint(f);
  unsigned int r = u + 0x7FFFu + ((u >> 16) & 1u);   // round-to-nearest-even
  return (unsigned short)(r >> 16);
}
__device__ __forceinline__ float bflo(unsigned int u) { return __uint_as_float(u << 16); }
__device__ __forceinline__ float bfhi(unsigned int u) { return __uint_as_float(u & 0xFFFF0000u); }

__device__ __forceinline__ float phi_elu1(float x) {   // elu(x)+1
  return x >= 0.f ? x + 1.f : __expf(x);
}

// A fragment (16x32 bf16, M x K): lane m=lane&15; K=0..15 for
// lanes 0-15 and K=16..31 for lanes 16-31 (matches B fragment).
__device__ __forceinline__ v16bf load_a_frag(const unsigned short* base, int lda) {
  const int lane = threadIdx.x & 31;
  const unsigned short* p = base + (size_t)(lane & 15) * lda + (lane >> 4) * 16;
  union { v16bf v; v8bf h[2]; } u;
  u.h[0] = *(const v8bf*)p;
  u.h[1] = *(const v8bf*)(p + 8);
  return u.v;
}
// B fragment (32x16 bf16, K x N) from a [N][K] (B^T row-major) buffer:
// lane n=lane&15 holds K=0..15 (lanes 0-15) or K=16..31 (lanes 16-31).
__device__ __forceinline__ v16bf load_b_frag(const unsigned short* base, int ldb) {
  const int lane = threadIdx.x & 31;
  return *(const v16bf*)(base + (size_t)(lane & 15) * ldb + (lane >> 4) * 16);
}
__device__ __forceinline__ v8f wmma_bf16(v16bf a, v16bf b, v8f c) {
  return __builtin_amdgcn_wmma_f32_16x16x32_bf16(false, a, false, b, (short)0, c, false, false);
}
__device__ __forceinline__ void sched_fence() {
#if __has_builtin(__builtin_amdgcn_sched_barrier)
  __builtin_amdgcn_sched_barrier(0);
#endif
}

// ---------------------------------------------------------------------------
// M=32768, N=1024, K=1024 GEMM: C = A @ Bm^T, Bm stored [N][K].
// block 256 = 8 waves (2x4), wave tile 64x64, block tile 128x256.
// MODE 0: phi, bf16 row-major out          (Q)
// MODE 1: phi, bf16 transposed-per-batch   (KT)
// MODE 2:      bf16 transposed-per-batch   (VT)
// MODE 3: +bias, fp32 row-major out        (final)
// ---------------------------------------------------------------------------
template<int MODE>
__global__ __launch_bounds__(256)
void gemm1024_kernel(const unsigned short* __restrict__ A,
                     const unsigned short* __restrict__ Bm,
                     unsigned short* __restrict__ obf,
                     float* __restrict__ of32,
                     const float* __restrict__ bias)
{
  const int lane = threadIdx.x & 31;
  const int wave = threadIdx.x >> 5;
  const int sel  = lane >> 4;
  const int ln   = lane & 15;
  const int rowbase = blockIdx.x * 128 + (wave >> 2) * 64;
  const int colbase = blockIdx.y * 256 + (wave & 3) * 64;

  v8f acc[4][4] = {};
  const unsigned short* Abase = A  + (size_t)rowbase * DDIM;
  const unsigned short* Bbase = Bm + (size_t)colbase * DDIM;

  for (int k = 0; k < DDIM; k += 32) {
    v16bf af[4], bfr[4];
#pragma unroll
    for (int mi = 0; mi < 4; ++mi)
      af[mi] = load_a_frag(Abase + (size_t)mi * 16 * DDIM + k, DDIM);
#pragma unroll
    for (int ni = 0; ni < 4; ++ni)
      bfr[ni] = load_b_frag(Bbase + (size_t)ni * 16 * DDIM + k, DDIM);
    {   // branchless stream-ahead prefetch of A (wraps at end; harmless)
      int kp = (k + 128) & (DDIM - 1);
#pragma unroll
      for (int mi = 0; mi < 4; ++mi)
        __builtin_prefetch(Abase + (size_t)mi * 16 * DDIM + kp +
                           (size_t)ln * DDIM, 0, 1);
    }
#pragma unroll
    for (int mi = 0; mi < 4; ++mi)
#pragma unroll
      for (int ni = 0; ni < 4; ++ni)
        acc[mi][ni] = wmma_bf16(af[mi], bfr[ni], acc[mi][ni]);
  }

  if constexpr (MODE == 3) {
#pragma unroll
    for (int mi = 0; mi < 4; ++mi)
#pragma unroll
      for (int ni = 0; ni < 4; ++ni)
#pragma unroll
        for (int r = 0; r < 8; ++r) {
          int m = rowbase + mi * 16 + r + 8 * sel;
          int n = colbase + ni * 16 + ln;
          of32[(size_t)m * DDIM + n] = acc[mi][ni][r] + bias[n];
        }
  } else {
    __shared__ unsigned short lds[8 * 32 * 72];   // 36 KB: 8 waves x 32 x 72
    unsigned short* w = lds + wave * (32 * 72);

    // Pack to bf16 (with activation) first: 128 live VGPRs -> 64.
    unsigned int pk[4][4][4];
#pragma unroll
    for (int mi = 0; mi < 4; ++mi)
#pragma unroll
      for (int ni = 0; ni < 4; ++ni)
#pragma unroll
        for (int r2 = 0; r2 < 4; ++r2) {
          float v0 = acc[mi][ni][2 * r2];
          float v1 = acc[mi][ni][2 * r2 + 1];
          if (MODE <= 1) { v0 = phi_elu1(v0); v1 = phi_elu1(v1); }
          pk[mi][ni][r2] = (unsigned int)f2bf(v0) | ((unsigned int)f2bf(v1) << 16);
        }

    for (int half = 0; half < 2; ++half) {
      if constexpr (MODE == 0) {      // rows [half*32,+32) row-major in LDS
#pragma unroll
        for (int mi2 = 0; mi2 < 2; ++mi2) {
          int mi = half * 2 + mi2;
#pragma unroll
          for (int ni = 0; ni < 4; ++ni)
#pragma unroll
            for (int r = 0; r < 8; ++r) {
              int m = mi * 16 + r + 8 * sel;
              int n = ni * 16 + ln;
              unsigned int v = pk[mi][ni][r >> 1] >> ((r & 1) * 16);
              w[(m - half * 32) * 72 + n] = (unsigned short)v;
            }
        }
      } else {                        // cols [half*32,+32) feature-major (transpose)
        unsigned int* w32 = (unsigned int*)w;
#pragma unroll
        for (int ni2 = 0; ni2 < 2; ++ni2) {
          int ni = half * 2 + ni2;
#pragma unroll
          for (int mi = 0; mi < 4; ++mi)
#pragma unroll
            for (int r2 = 0; r2 < 4; ++r2) {
              int m = mi * 16 + 2 * r2 + 8 * sel;   // even; pair (m, m+1)
              int n = ni * 16 + ln;
              w32[(n - half * 32) * 36 + (m >> 1)] = pk[mi][ni][r2];
            }
        }
      }
      __syncthreads();
#pragma unroll
      for (int it = 0; it < 8; ++it) {
        int line = it * 4 + (lane >> 3);
        int eo   = (lane & 7) * 8;
        v8bf d = *(const v8bf*)(w + line * 72 + eo);
        if constexpr (MODE == 0) {
          *(v8bf*)(obf + (size_t)(rowbase + half * 32 + line) * DDIM + colbase + eo) = d;
        } else {
          int f = colbase + half * 32 + line;       // global feature
          int b = rowbase >> 13;                    // batch (8192 rows/batch)
          int t = (rowbase & (NTOK - 1)) + eo;      // token within batch
          *(v8bf*)(obf + ((size_t)b * DDIM + f) * NTOK + t) = d;
        }
      }
      __syncthreads();
    }
  }
}

// ---------------------------------------------------------------------------
// kvT[b,h] += v^T k over a K-split of the token range.  grid (64, 16),
// block 64 = 2 waves, wave tile 32x64 (2 A-strips x 4 B-tiles, 8 WMMA/step).
// Every fragment now has >=2 consumers, and a sched_barrier pins the 12
// loads ahead of the WMMA group -> partial s_wait_loadcnt, real overlap.
// ---------------------------------------------------------------------------
__global__ __launch_bounds__(64)
void kv_kernel(const unsigned short* __restrict__ VT,
               const unsigned short* __restrict__ KT,
               float* __restrict__ kvacc)
{
  const int lane = threadIdx.x & 31;
  const int wave = threadIdx.x >> 5;          // e-half 0..1
  const int sel = lane >> 4, ln = lane & 15;
  const int bh = blockIdx.x;                  // 0..63
  const int b = bh >> 4, h = bh & 15;
  const int kbase = blockIdx.y * (NTOK / KVSPLIT);

  const unsigned short* Abase = VT + ((size_t)b * DDIM + h * HD + wave * 32) * NTOK + kbase;
  const unsigned short* Bbase = KT + ((size_t)b * DDIM + h * HD) * NTOK + kbase;

  v8f acc[2][4] = {};
  for (int k = 0; k < NTOK / KVSPLIT; k += 32) {
    v16bf a0 = load_a_frag(Abase + k, NTOK);
    v16bf a1 = load_a_frag(Abase + (size_t)16 * NTOK + k, NTOK);
    v16bf bfr[4];
#pragma unroll
    for (int ni = 0; ni < 4; ++ni)
      bfr[ni] = load_b_frag(Bbase + (size_t)ni * 16 * NTOK + k, NTOK);
    sched_fence();                            // loads stay ahead of WMMAs
#pragma unroll
    for (int ni = 0; ni < 4; ++ni) {
      acc[0][ni] = wmma_bf16(a0, bfr[ni], acc[0][ni]);
      acc[1][ni] = wmma_bf16(a1, bfr[ni], acc[1][ni]);
    }
  }
  float* dst = kvacc + (size_t)bh * (HD * HD);
#pragma unroll
  for (int st = 0; st < 2; ++st)
#pragma unroll
    for (int ni = 0; ni < 4; ++ni)
#pragma unroll
      for (int r = 0; r < 8; ++r) {
        int e  = wave * 32 + st * 16 + r + 8 * sel;
        int dd = ni * 16 + ln;
        atomicAdd(dst + e * HD + dd, acc[st][ni][r]);
      }
}

// ksum row (row 64 of kvp) straight from KT.  grid 512 x 256, 1 wave / feature.
__global__ __launch_bounds__(256)
void ksum_kernel(const unsigned short* __restrict__ KT,
                 unsigned short* __restrict__ kvp)
{
  const int lane = threadIdx.x & 31, wave = threadIdx.x >> 5;
  const int gr = blockIdx.x * 8 + wave;        // (b, f) row of KT, 0..4095
  const unsigned short* p = KT + (size_t)gr * NTOK;
  float s = 0.f;
  for (int i = lane * 8; i < NTOK; i += 256) {
    uint4 v = *(const uint4*)(p + i);
    s += bflo(v.x) + bfhi(v.x) + bflo(v.y) + bfhi(v.y)
       + bflo(v.z) + bfhi(v.z) + bflo(v.w) + bfhi(v.w);
  }
#pragma unroll
  for (int off = 16; off > 0; off >>= 1) s += __shfl_down(s, off, 32);
  if (lane == 0) {
    int b = gr >> 10, f = gr & 1023;
    int bh = b * NH + (f >> 6), dd = f & 63;
    kvp[(size_t)bh * (80 * HD) + 64 * HD + dd] = f2bf(s);
  }
}

__global__ void kv_finalize_kernel(const float* __restrict__ kvacc,
                                   unsigned short* __restrict__ kvp)
{
  int idx = blockIdx.x * blockDim.x + threadIdx.x;   // over 64*80*64
  if (idx >= 64 * 80 * 64) return;
  int bh = idx / (80 * HD);
  int rem = idx % (80 * HD);
  int row = rem / HD, col = rem % HD;
  if (row < 64)      kvp[idx] = f2bf(kvacc[(size_t)bh * 4096 + row * HD + col]);
  else if (row > 64) kvp[idx] = 0;                   // row 64 written by ksum
}

// numerator + fused divide: y = (q @ kvp) / (col64 + eps).  grid (64, 16).
__global__ __launch_bounds__(256)
void numer_kernel(const unsigned short* __restrict__ Q,
                  const unsigned short* __restrict__ kvp,
                  unsigned short* __restrict__ y)
{
  const int lane = threadIdx.x & 31, wave = threadIdx.x >> 5;
  const int sel = lane >> 4, ln = lane & 15;
  const int h = blockIdx.y;
  const int rowbase = blockIdx.x * 512 + wave * 64;
  const int bh = (rowbase >> 13) * NH + h;

  const unsigned short* Abase = Q   + (size_t)rowbase * DDIM + h * HD;
  const unsigned short* Bbase = kvp + (size_t)bh * (80 * HD);

  v8f acc[4][5] = {};
#pragma unroll
  for (int k = 0; k < 64; k += 32) {
    v16bf af[4], bfr[5];
#pragma unroll
    for (int mi = 0; mi < 4; ++mi)
      af[mi] = load_a_frag(Abase + (size_t)mi * 16 * DDIM + k, DDIM);
#pragma unroll
    for (int ni = 0; ni < 5; ++ni)
      bfr[ni] = load_b_frag(Bbase + ni * 16 * HD + k, HD);
#pragma unroll
    for (int mi = 0; mi < 4; ++mi)
#pragma unroll
      for (int ni = 0; ni < 5; ++ni)
        acc[mi][ni] = wmma_bf16(af[mi], bfr[ni], acc[mi][ni]);
  }
#pragma unroll
  for (int mi = 0; mi < 4; ++mi)
#pragma unroll
    for (int r = 0; r < 8; ++r) {
      float denom = __shfl(acc[mi][4][r], lane & 16, 32) + 1e-6f;  // bcast col 64
      int m = rowbase + mi * 16 + r + 8 * sel;
#pragma unroll
      for (int ni = 0; ni < 4; ++ni)
        y[(size_t)m * DDIM + h * HD + ni * 16 + ln] = f2bf(acc[mi][ni][r] / denom);
    }
}

__global__ void f32_to_bf16_kernel(const float* __restrict__ src,
                                   unsigned short* __restrict__ dst, int n)
{
  int i = (blockIdx.x * blockDim.x + threadIdx.x) * 4;
  if (i + 3 < n) {
    float4 v = *(const float4*)(src + i);
    unsigned long long pk = (unsigned long long)f2bf(v.x)
                          | ((unsigned long long)f2bf(v.y) << 16)
                          | ((unsigned long long)f2bf(v.z) << 32)
                          | ((unsigned long long)f2bf(v.w) << 48);
    *(unsigned long long*)(dst + i) = pk;
  }
}

__global__ void zero_f32_kernel(float* __restrict__ p, int n)
{
  int i = blockIdx.x * blockDim.x + threadIdx.x;
  if (i < n) p[i] = 0.f;
}

// ---------------------------------------------------------------------------
extern "C" void kernel_launch(void* const* d_in, const int* in_sizes, int n_in,
                              void* d_out, int out_size, void* d_ws, size_t ws_size,
                              hipStream_t stream) {
  const float* x  = (const float*)d_in[0];
  const float* Wq = (const float*)d_in[1];
  const float* Wk = (const float*)d_in[2];
  const float* Wv = (const float*)d_in[3];
  const float* Wo = (const float*)d_in[4];
  const float* bo = (const float*)d_in[5];
  (void)in_sizes; (void)n_in; (void)out_size; (void)ws_size;

  char* ws = (char*)d_ws;
  size_t off = 0;
  auto alloc = [&](size_t bytes) {
    char* p = ws + off;
    off = (off + bytes + 255) & ~(size_t)255;
    return p;
  };
  unsigned short* xb  = (unsigned short*)alloc((size_t)MTOT * DDIM * 2);  // reused as y
  unsigned short* qb  = (unsigned short*)alloc((size_t)MTOT * DDIM * 2);
  unsigned short* ktb = (unsigned short*)alloc((size_t)MTOT * DDIM * 2);
  unsigned short* vtb = (unsigned short*)alloc((size_t)MTOT * DDIM * 2);
  unsigned short* wqb = (unsigned short*)alloc((size_t)DDIM * DDIM * 2);
  unsigned short* wkb = (unsigned short*)alloc((size_t)DDIM * DDIM * 2);
  unsigned short* wvb = (unsigned short*)alloc((size_t)DDIM * DDIM * 2);
  unsigned short* wob = (unsigned short*)alloc((size_t)DDIM * DDIM * 2);
  float*          kvacc = (float*)alloc((size_t)64 * HD * HD * 4);
  unsigned short* kvp   = (unsigned short*)alloc((size_t)64 * 80 * HD * 2);
  unsigned short* yb = xb;   // x(bf16) dead after the three projection GEMMs

  // 1) precision conversion
  f32_to_bf16_kernel<<<(MTOT * DDIM / 4 + 255) / 256, 256, 0, stream>>>(x,  xb,  MTOT * DDIM);
  f32_to_bf16_kernel<<<(DDIM * DDIM / 4 + 255) / 256, 256, 0, stream>>>(Wq, wqb, DDIM * DDIM);
  f32_to_bf16_kernel<<<(DDIM * DDIM / 4 + 255) / 256, 256, 0, stream>>>(Wk, wkb, DDIM * DDIM);
  f32_to_bf16_kernel<<<(DDIM * DDIM / 4 + 255) / 256, 256, 0, stream>>>(Wv, wvb, DDIM * DDIM);
  f32_to_bf16_kernel<<<(DDIM * DDIM / 4 + 255) / 256, 256, 0, stream>>>(Wo, wob, DDIM * DDIM);
  zero_f32_kernel<<<(64 * 4096 + 255) / 256, 256, 0, stream>>>(kvacc, 64 * 4096);

  // 2) projections
  dim3 ggrid(MTOT / 128, DDIM / 256);
  gemm1024_kernel<0><<<ggrid, 256, 0, stream>>>(xb, wqb, qb,  nullptr, nullptr);
  gemm1024_kernel<1><<<ggrid, 256, 0, stream>>>(xb, wkb, ktb, nullptr, nullptr);
  gemm1024_kernel<2><<<ggrid, 256, 0, stream>>>(xb, wvb, vtb, nullptr, nullptr);

  // 3) per-head state
  kv_kernel<<<dim3(64, KVSPLIT), 64, 0, stream>>>(vtb, ktb, kvacc);
  ksum_kernel<<<512, 256, 0, stream>>>(ktb, kvp);
  kv_finalize_kernel<<<(64 * 80 * HD + 255) / 256, 256, 0, stream>>>(kvacc, kvp);

  // 4) numerator/denominator fused
  numer_kernel<<<dim3(MTOT / 512, NH), 256, 0, stream>>>(qb, kvp, yb);

  // 5) output projection + bias
  gemm1024_kernel<3><<<ggrid, 256, 0, stream>>>(yb, wob, nullptr, (float*)d_out, bo);
}
